// ModelLayer_50869592655494
// MI455X (gfx1250) — matrix operand order repaired and checked by reading, not compiled, not run on hardware
//
#include <hip/hip_runtime.h>
#include <hip/hip_bf16.h>
#include <stdint.h>

// ---------------------------------------------------------------------------
// Problem constants (match the reference)
// ---------------------------------------------------------------------------
#define HCH   128
#define NNODE 100000
#define NEDGE 200000
#define NC5   30000
#define NC6   30000
#define R5    (5 * NC5)   // 150000 cycle5 rows
#define R6    (6 * NC6)   // 180000 cycle6 rows
#define RTOT  (R5 + R6)   // 330000 combined cycle rows

typedef __attribute__((ext_vector_type(16))) _Float16 v16h;
typedef __attribute__((ext_vector_type(8)))  float    v8f;

union V16H { v16h h; uint32_t u[8]; };

__device__ __forceinline__ void row_decode(int r, const int* __restrict__ c5,
                                           const int* __restrict__ c6,
                                           const int*& row, int& L, int& j) {
  if (r < R5) { int c = r / 5; j = r - c * 5; L = 5; row = c5 + c * 5; }
  else        { int rr = r - R5; int c = rr / 6; j = rr - c * 6; L = 6; row = c6 + c * 6; }
}

// ---------------------------------------------------------------------------
// Generic f16 WMMA GEMM with fused epilogues.
//   A[MxK] f16 row-major, Bt[n*K+k] f16 (pre-transposed weight), Nc = NT*16.
//   MODE 0: C32[m*Nc+n]              = relu(acc)          (fp32 store)
//   MODE 1: C16[m*c16s + ncol0 + n]  = (f16)relu(acc)     (operand-build store)
//   MODE 2: relu(acc) atomically scattered: row m -> edges row[j], row[j-1]
//           (cycle_to_edge), Nc must be 128.
// One wave computes a 16-row strip x full Nc.  All M here are multiples of 16.
// ---------------------------------------------------------------------------
template <int NT, int MODE>
__global__ __launch_bounds__(128) void wmma_gemm(
    const _Float16* __restrict__ A, const _Float16* __restrict__ Bt,
    float* __restrict__ C32, _Float16* __restrict__ C16, int ncol0, int c16s,
    const int* __restrict__ c5, const int* __restrict__ c6,
    float* __restrict__ scat, int M, int K) {
  const int lane   = threadIdx.x & 31;
  const int wave   = threadIdx.x >> 5;
  const int strip  = blockIdx.x * 4 + wave;
  const int strips = M >> 4;
  if (strip >= strips) return;
  const int m0  = strip << 4;
  const int l15 = lane & 15;
  const int hid = lane >> 4;   // which half of the wave

  v8f acc[NT] = {};

  const uint32_t* Arow = (const uint32_t*)(A + (size_t)(m0 + l15) * K);

  for (int k0 = 0; k0 < K; k0 += 32) {
    // A fragment: 16x32 f16, contiguous K (lanes 0-15: K 0-15, 16-31: K 16-31)
    V16H a;
    const int ab = (k0 >> 1) + hid * 8;
#pragma unroll
    for (int v = 0; v < 4; ++v) a.u[v] = Arow[ab + v];
#pragma unroll
    for (int v = 0; v < 4; ++v) a.u[4 + v] = Arow[ab + 4 + v];

#pragma unroll
    for (int nt = 0; nt < NT; ++nt) {
      const int n = (nt << 4) + l15;
      const uint32_t* Brow = (const uint32_t*)(Bt + (size_t)n * K);
      V16H b;
      const int bb = (k0 >> 1) + hid * 8;
#pragma unroll
      for (int v = 0; v < 8; ++v) b.u[v] = Brow[bb + v];
      acc[nt] = __builtin_amdgcn_wmma_f32_16x16x32_f16(
          false, a.h, false, b.h, (short)0, acc[nt], false, false);
    }
  }

  const int Nc = NT << 4;
  if (MODE == 0) {
#pragma unroll
    for (int nt = 0; nt < NT; ++nt) {
      const int n = (nt << 4) + l15;
#pragma unroll
      for (int v = 0; v < 8; ++v) {
        const int m = m0 + hid * 8 + v;
        float x = acc[nt][v];
        C32[(size_t)m * Nc + n] = x > 0.f ? x : 0.f;
      }
    }
  } else if (MODE == 1) {
#pragma unroll
    for (int nt = 0; nt < NT; ++nt) {
      const int n = (nt << 4) + l15;
#pragma unroll
      for (int v = 0; v < 8; ++v) {
        const int m = m0 + hid * 8 + v;
        float x = acc[nt][v];
        C16[(size_t)m * c16s + ncol0 + n] = (_Float16)(x > 0.f ? x : 0.f);
      }
    }
  } else {  // MODE == 2: cycle_to_edge atomic scatter, channel = n (Nc==128)
#pragma unroll
    for (int v = 0; v < 8; ++v) {
      const int m = m0 + hid * 8 + v;
      const int* row; int L, j;
      row_decode(m, c5, c6, row, L, j);
      const int jp = (j == 0) ? (L - 1) : (j - 1);
      const size_t e0 = (size_t)row[j]  * 128;
      const size_t e1 = (size_t)row[jp] * 128;
#pragma unroll
      for (int nt = 0; nt < NT; ++nt) {
        const int n = (nt << 4) + l15;
        float x = acc[nt][v];
        x = x > 0.f ? x : 0.f;
        atomicAdd(&scat[e0 + n], x);
        atomicAdd(&scat[e1 + n], x);
      }
    }
  }
}

// ---------------------------------------------------------------------------
// Elementwise / gather / scatter kernels
// ---------------------------------------------------------------------------
__global__ void k_zero(float* __restrict__ p, long long n) {
  long long i = (long long)blockIdx.x * blockDim.x + threadIdx.x;
  if (i < n) p[i] = 0.f;
}

// Bt[n*K+k] = (f16) W[k*Nc+n]
__global__ void k_w_transpose(const float* __restrict__ W,
                              _Float16* __restrict__ Bt, int K, int Nc) {
  int i = blockIdx.x * blockDim.x + threadIdx.x;
  if (i >= K * Nc) return;
  int k = i / Nc, n = i - k * Nc;
  Bt[(size_t)n * K + k] = (_Float16)W[i];
}

// lift_ne = node[u]+node[v]; A1 = f16 concat(lift_ne, edge_rep)  (E x 256)
__global__ void k_lift_ne(const float* __restrict__ node,
                          const float* __restrict__ edge,
                          const int* __restrict__ en,
                          float* __restrict__ lift_ne,
                          _Float16* __restrict__ A1) {
  long long i = (long long)blockIdx.x * blockDim.x + threadIdx.x;
  if (i >= (long long)NEDGE * HCH) return;
  int e = (int)(i / HCH), c = (int)(i - (long long)e * HCH);
  int u = en[2 * e], v = en[2 * e + 1];
  float l = node[(size_t)u * HCH + c] + node[(size_t)v * HCH + c];
  lift_ne[i] = l;
  A1[(size_t)e * 256 + c]       = (_Float16)l;
  A1[(size_t)e * 256 + 128 + c] = (_Float16)edge[i];
}

// lvl_ne[u] += edge_h; lvl_ne[v] += edge_h
__global__ void k_scatter_lvlne(const float* __restrict__ edge_h,
                                const int* __restrict__ en,
                                float* __restrict__ lvl_ne) {
  long long i = (long long)blockIdx.x * blockDim.x + threadIdx.x;
  if (i >= (long long)NEDGE * HCH) return;
  int e = (int)(i / HCH), c = (int)(i - (long long)e * HCH);
  float x = edge_h[i];
  atomicAdd(&lvl_ne[(size_t)en[2 * e] * HCH + c], x);
  atomicAdd(&lvl_ne[(size_t)en[2 * e + 1] * HCH + c], x);
}

// A2 = f16((1+eps1)*node + lvl_ne)   (N x 128)
__global__ void k_prep_a2(const float* __restrict__ node,
                          const float* __restrict__ lvl_ne,
                          const float* __restrict__ eps1,
                          _Float16* __restrict__ A2) {
  long long i = (long long)blockIdx.x * blockDim.x + threadIdx.x;
  if (i >= (long long)NNODE * HCH) return;
  A2[i] = (_Float16)((1.0f + *eps1) * node[i] + lvl_ne[i]);
}

// A3 = f16((1+eps2)*edge_h + lift_ne)  (E x 128)
__global__ void k_prep_a3(const float* __restrict__ edge_h,
                          const float* __restrict__ lift_ne,
                          const float* __restrict__ eps2,
                          _Float16* __restrict__ A3) {
  long long i = (long long)blockIdx.x * blockDim.x + threadIdx.x;
  if (i >= (long long)NEDGE * HCH) return;
  A3[i] = (_Float16)((1.0f + *eps2) * edge_h[i] + lift_ne[i]);
}

// A4 (f16, RTOT x 512) = concat(g11(edge_to_cycle(edge_rep)), cycle_rep)
__global__ void k_lift_cycle(const float* __restrict__ edge,
                             const float* __restrict__ crep,
                             const int* __restrict__ c5,
                             const int* __restrict__ c6,
                             _Float16* __restrict__ A4) {
  long long i = (long long)blockIdx.x * blockDim.x + threadIdx.x;
  if (i >= (long long)RTOT * HCH) return;
  int r = (int)(i / HCH), ch = (int)(i - (long long)r * HCH);
  const int* row; int L, j;
  row_decode(r, c5, c6, row, L, j);
  float s = 0.f;
  for (int t = 0; t < L; ++t) s += edge[(size_t)row[t] * HCH + ch];
  int jp = (j == 0) ? (L - 1) : (j - 1);
  float x = edge[(size_t)row[j] * HCH + ch] + edge[(size_t)row[jp] * HCH + ch];
  float s2 = 2.f * s;   // sum over positions of (ev[j]+ev[j-1])
  A4[(size_t)r * 512 + ch]       = (_Float16)x;
  A4[(size_t)r * 512 + 128 + ch] = (_Float16)s2;
  A4[(size_t)r * 512 + 256 + ch] = (_Float16)crep[(size_t)r * 256 + ch];
  A4[(size_t)r * 512 + 384 + ch] = (_Float16)crep[(size_t)r * 256 + 128 + ch];
}

// cycle_to_edge scatter: src[r] adds to edges row[j] and row[j-1]  (CH channels)
__global__ void k_scatter_c2e(const float* __restrict__ src,
                              const int* __restrict__ c5,
                              const int* __restrict__ c6,
                              float* __restrict__ out, int CH) {
  long long i = (long long)blockIdx.x * blockDim.x + threadIdx.x;
  if (i >= (long long)RTOT * CH) return;
  int r = (int)(i / CH), ch = (int)(i - (long long)r * CH);
  const int* row; int L, j;
  row_decode(r, c5, c6, row, L, j);
  int jp = (j == 0) ? (L - 1) : (j - 1);
  float x = src[i];
  atomicAdd(&out[(size_t)row[j]  * CH + ch], x);
  atomicAdd(&out[(size_t)row[jp] * CH + ch], x);
}

// A5 = f16((1+e11)*edge_rep + (1+e12)*lvl_ec)   (E x 128)
__global__ void k_prep_a5(const float* __restrict__ edge,
                          const float* __restrict__ lvl_ec,
                          const float* __restrict__ e11,
                          const float* __restrict__ e12,
                          _Float16* __restrict__ A5) {
  long long i = (long long)blockIdx.x * blockDim.x + threadIdx.x;
  if (i >= (long long)NEDGE * HCH) return;
  A5[i] = (_Float16)((1.0f + *e11) * edge[i] + (1.0f + *e12) * lvl_ec[i]);
}

// lin = edge_to_cycle(intermediate);  A6 = f16((1+e2)*lin + lift_f16)
// lift read back as f16 from A4's first 256 columns.  (RTOT x 256)
__global__ void k_prep_a6(const float* __restrict__ inter,
                          const _Float16* __restrict__ A4,
                          const int* __restrict__ c5,
                          const int* __restrict__ c6,
                          const float* __restrict__ e2,
                          _Float16* __restrict__ A6) {
  long long i = (long long)blockIdx.x * blockDim.x + threadIdx.x;
  if (i >= (long long)RTOT * 256) return;
  int r = (int)(i / 256), ch = (int)(i - (long long)r * 256);
  const int* row; int L, j;
  row_decode(r, c5, c6, row, L, j);
  int jp = (j == 0) ? (L - 1) : (j - 1);
  float lin = inter[(size_t)row[j] * 256 + ch] + inter[(size_t)row[jp] * 256 + ch];
  float lift = (float)A4[(size_t)r * 512 + ch];
  A6[i] = (_Float16)((1.0f + *e2) * lin + lift);
}

// ---------------------------------------------------------------------------
// Host launch
// ---------------------------------------------------------------------------
static inline long long cdivll(long long a, long long b) { return (a + b - 1) / b; }

extern "C" void kernel_launch(void* const* d_in, const int* in_sizes, int n_in,
                              void* d_out, int out_size, void* d_ws, size_t ws_size,
                              hipStream_t stream) {
  (void)in_sizes; (void)n_in; (void)out_size; (void)ws_size;
  const float* node_rep  = (const float*)d_in[0];
  const float* edge_rep  = (const float*)d_in[1];
  const float* cycle_rep = (const float*)d_in[2];
  const float* ne_lift_W = (const float*)d_in[3];
  const float* ne_l1_W   = (const float*)d_in[4];
  const float* ne_l2_W   = (const float*)d_in[5];
  const float* ec_lift_W = (const float*)d_in[6];
  const float* ec_l1_W   = (const float*)d_in[7];
  const float* ec_l2_W   = (const float*)d_in[8];
  const float* mlp_W     = (const float*)d_in[9];
  const float* ne_eps1   = (const float*)d_in[10];
  const float* ne_eps2   = (const float*)d_in[11];
  const float* ec_e11    = (const float*)d_in[12];
  const float* ec_e12    = (const float*)d_in[13];
  const float* ec_e2     = (const float*)d_in[14];
  const int*   edge_nodes = (const int*)d_in[15];
  const int*   cyc5       = (const int*)d_in[16];
  const int*   cyc6       = (const int*)d_in[17];

  float* out       = (float*)d_out;
  float* node_out  = out;                                   // N x 128
  float* edge_out  = out + (size_t)NNODE * HCH;             // E x 128
  float* cycle_out = edge_out + (size_t)NEDGE * HCH;        // RTOT x 256

  // Workspace layout
  char* w = (char*)d_ws;
  size_t off = 0;
  auto alloc = [&](size_t bytes) {
    void* p = w + off;
    off = (off + bytes + 255) & ~(size_t)255;
    return p;
  };
  _Float16* Bt1 = (_Float16*)alloc((size_t)256 * 128 * 2);  // ne_l1
  _Float16* Bt2 = (_Float16*)alloc((size_t)128 * 128 * 2);  // ne_l2
  _Float16* Bt3 = (_Float16*)alloc((size_t)128 * 128 * 2);  // ne_lift
  _Float16* Bt4 = (_Float16*)alloc((size_t)512 * 128 * 2);  // ec_l1
  _Float16* Bt5 = (_Float16*)alloc((size_t)128 * 128 * 2);  // ec_l2
  _Float16* Bt6 = (_Float16*)alloc((size_t)256 * 256 * 2);  // ec_lift
  _Float16* Bt7 = (_Float16*)alloc((size_t)256 * 128 * 2);  // mlp
  float* lift_ne = (float*)alloc((size_t)NEDGE * HCH * 4);
  float* edge_h  = (float*)alloc((size_t)NEDGE * HCH * 4);
  float* lvl_ne  = (float*)alloc((size_t)NNODE * HCH * 4);
  float* inter   = (float*)alloc((size_t)NEDGE * 256 * 4);
  float* lvl_ec  = (float*)alloc((size_t)NEDGE * HCH * 4);
  _Float16* Aun  = (_Float16*)alloc((size_t)NEDGE * 256 * 2); // union A1/A2/A3/A5
  _Float16* A4   = (_Float16*)alloc((size_t)RTOT * 512 * 2);
  _Float16* A6   = (_Float16*)alloc((size_t)RTOT * 256 * 2);
  _Float16* A7   = (_Float16*)alloc((size_t)NEDGE * 256 * 2);

  const int TB = 256;
  auto g_f32 = [&](const _Float16* A, const _Float16* Bt, float* C, int M, int K, bool wide) {
    dim3 g((unsigned)cdivll(M >> 4, 4));
    if (wide) wmma_gemm<16, 0><<<g, 128, 0, stream>>>(A, Bt, C, nullptr, 0, 0, nullptr, nullptr, nullptr, M, K);
    else      wmma_gemm<8,  0><<<g, 128, 0, stream>>>(A, Bt, C, nullptr, 0, 0, nullptr, nullptr, nullptr, M, K);
  };
  auto g_f16 = [&](const _Float16* A, const _Float16* Bt, _Float16* C, int ncol0, int M, int K) {
    dim3 g((unsigned)cdivll(M >> 4, 4));
    wmma_gemm<8, 1><<<g, 128, 0, stream>>>(A, Bt, nullptr, C, ncol0, 256, nullptr, nullptr, nullptr, M, K);
  };
  auto g_scat = [&](const _Float16* A, const _Float16* Bt, float* scat, int M, int K) {
    dim3 g((unsigned)cdivll(M >> 4, 4));
    wmma_gemm<8, 2><<<g, 128, 0, stream>>>(A, Bt, nullptr, nullptr, 0, 0, cyc5, cyc6, scat, M, K);
  };

  // 0) weights -> f16 transposed
  k_w_transpose<<<(unsigned)cdivll(256 * 128, TB), TB, 0, stream>>>(ne_l1_W,  Bt1, 256, 128);
  k_w_transpose<<<(unsigned)cdivll(128 * 128, TB), TB, 0, stream>>>(ne_l2_W,  Bt2, 128, 128);
  k_w_transpose<<<(unsigned)cdivll(128 * 128, TB), TB, 0, stream>>>(ne_lift_W, Bt3, 128, 128);
  k_w_transpose<<<(unsigned)cdivll(512 * 128, TB), TB, 0, stream>>>(ec_l1_W,  Bt4, 512, 128);
  k_w_transpose<<<(unsigned)cdivll(128 * 128, TB), TB, 0, stream>>>(ec_l2_W,  Bt5, 128, 128);
  k_w_transpose<<<(unsigned)cdivll(256 * 256, TB), TB, 0, stream>>>(ec_lift_W, Bt6, 256, 256);
  k_w_transpose<<<(unsigned)cdivll(256 * 128, TB), TB, 0, stream>>>(mlp_W,    Bt7, 256, 128);

  const long long nEH = (long long)NEDGE * HCH;
  const long long nRH = (long long)RTOT * HCH;

  // 1) lift_ne + A1;  G1: edge_h = relu([lift_ne, edge_rep] @ ne_l1_W)
  k_lift_ne<<<(unsigned)cdivll(nEH, TB), TB, 0, stream>>>(node_rep, edge_rep, edge_nodes, lift_ne, Aun);
  g_f32(Aun, Bt1, edge_h, NEDGE, 256, false);

  // 2) lvl_ne scatter;  G2: node_out
  k_zero<<<(unsigned)cdivll((long long)NNODE * HCH, TB), TB, 0, stream>>>(lvl_ne, (long long)NNODE * HCH);
  k_scatter_lvlne<<<(unsigned)cdivll(nEH, TB), TB, 0, stream>>>(edge_h, edge_nodes, lvl_ne);
  k_prep_a2<<<(unsigned)cdivll((long long)NNODE * HCH, TB), TB, 0, stream>>>(node_rep, lvl_ne, ne_eps1, Aun);
  g_f32(Aun, Bt2, node_out, NNODE, 128, false);

  // 3) G3: eo1 = relu(((1+eps2)*edge_h + lift_ne) @ ne_lift_W) -> A7 cols 0-127 (f16)
  k_prep_a3<<<(unsigned)cdivll(nEH, TB), TB, 0, stream>>>(edge_h, lift_ne, ne_eps2, Aun);
  g_f16(Aun, Bt3, A7, 0, NEDGE, 128);

  // 4) cycle lift + A4;  G4: relu([lift, cyc_rep] @ ec_l1_W) scattered into lvl_ec
  k_zero<<<(unsigned)cdivll(nEH, TB), TB, 0, stream>>>(lvl_ec, nEH);
  k_lift_cycle<<<(unsigned)cdivll(nRH, TB), TB, 0, stream>>>(edge_rep, cycle_rep, cyc5, cyc6, A4);
  g_scat(A4, Bt4, lvl_ec, RTOT, 512);

  // 5) intermediate = cycle_to_edge(cycle_rep) (256 ch)
  k_zero<<<(unsigned)cdivll((long long)NEDGE * 256, TB), TB, 0, stream>>>(inter, (long long)NEDGE * 256);
  k_scatter_c2e<<<(unsigned)cdivll((long long)RTOT * 256, TB), TB, 0, stream>>>(cycle_rep, cyc5, cyc6, inter, 256);

  // 6) G5: eo2 = relu(((1+e11)*edge_rep + (1+e12)*lvl_ec) @ ec_l2_W) -> A7 cols 128-255
  k_prep_a5<<<(unsigned)cdivll(nEH, TB), TB, 0, stream>>>(edge_rep, lvl_ec, ec_e11, ec_e12, Aun);
  g_f16(Aun, Bt5, A7, 128, NEDGE, 128);

  // 7) G6: cycle_out = relu(((1+e2)*lin + lift) @ ec_lift_W)   (Nc = 256)
  k_prep_a6<<<(unsigned)cdivll((long long)RTOT * 256, TB), TB, 0, stream>>>(inter, A4, cyc5, cyc6, ec_e2, A6);
  g_f32(A6, Bt6, cycle_out, RTOT, 256, true);

  // 8) G7: edge_out = relu([eo1, eo2] @ mlp_W)
  g_f32(A7, Bt7, edge_out, NEDGE, 256, false);
}
